// VoxelProposalLayerWarp_69260642615439
// MI455X (gfx1250) — compile-verified, matrix operand order checked by reading, not run on hardware
//
#include <hip/hip_runtime.h>
#include <hip/hip_bf16.h>

// ---------------- problem constants ----------------
#define EMBED   128
#define HEADS   8
#define LEVELS  3
#define POINTS  4
#define LAYERS  3
#define SX      128
#define SY      128
#define SZ      16
#define NVOX    (SX * SY * SZ)        // 262144
#define KMAX    32768
#define FH0     48
#define FW0     160
#define FH1     24
#define FW1     80
#define FH2     12
#define FW2     40
#define HW0     (FH0 * FW0)           // 7680
#define HW1     (FH1 * FW1)           // 1920
#define HW2     (FH2 * FW2)           // 480
#define VROWS   (HW0 + HW1 + HW2)     // 10080  (multiple of 16)
#define FFN     (4 * EMBED)           // 512

typedef __attribute__((ext_vector_type(16))) _Float16     v16h;
typedef __attribute__((ext_vector_type(8)))  float        v8f;
typedef __attribute__((ext_vector_type(8)))  unsigned int v8u;

#if __has_builtin(__builtin_amdgcn_global_load_async_to_lds_b128) && \
    __has_builtin(__builtin_amdgcn_s_wait_asynccnt)
#define HAVE_ASYNC_LDS 1
typedef int v4i __attribute__((vector_size(4 * sizeof(int))));
typedef __attribute__((address_space(1))) v4i* g1_v4i_p;
typedef __attribute__((address_space(3))) v4i* l3_v4i_p;
#else
#define HAVE_ASYNC_LDS 0
#endif

// =====================================================================
//  WMMA GEMM:  C[M,N] = A[M,K] (f16, row-major) x BT[N,K]^T (f16) + epi
//  - wave tile 64(M) x 32(N): 8 f32 accumulators, A frag reused x2,
//    B frag reused x4 (0.75 fragment loads per WMMA)
//  - B panel staged through LDS in 64-wide K slabs with async copies
//    (ASYNCcnt) when available; ds_load feeds B fragments.
//  Fragment layout per CDNA5 ISA 7.12.2 (16-bit A 16x32):
//    lane 0-15 : K in {0..7, 16..23}; lane 16-31: K in {8..15, 24..31}
//  EPI: 0 = bias -> f32 ; 1 = bias + residual -> f32 ; 2 = bias + GELU -> f16
// =====================================================================
template <int EPI>
__global__ __launch_bounds__(256) void gemm_wmma_f16(
    const _Float16* __restrict__ A,   // [M,K]
    const _Float16* __restrict__ BT,  // [N,K]
    const float* __restrict__ bias,   // [N]
    const float* __restrict__ resid,  // [M,N] or nullptr
    float* __restrict__ Cf,           // f32 out (EPI 0/1)
    _Float16* __restrict__ Ch,        // f16 out (EPI 2)
    int M, int Nn, int Kk)
{
    extern __shared__ __align__(16) _Float16 ldsB[];   // [Nn][64] halves per slab
    const int tid  = threadIdx.x;
    const int lane = tid & 31;
    const int wave = tid >> 5;
    const int l15  = lane & 15;
    const int hi   = (lane >> 4) << 2;                 // dword offset 0 or 4

    const int tilesN = Nn >> 5;                        // 32-wide N tiles
    const int tilesM = (M + 63) >> 6;                  // 64-tall M tiles
    const int tile   = blockIdx.x * 8 + wave;
    const bool active = tile < tilesM * tilesN;
    const int tm64 = active ? (tile / tilesN) << 6 : 0;
    const int tn32 = active ? (tile % tilesN) << 5 : 0;

    // A fragment row pointers (clamped for the M remainder tile)
    const unsigned int* apt[4];
#pragma unroll
    for (int mi = 0; mi < 4; ++mi) {
        int r = tm64 + mi * 16 + l15;
        if (r > M - 1) r = M - 1;
        apt[mi] = (const unsigned int*)(A + (size_t)r * Kk);
    }

    v8f acc[8];
#pragma unroll
    for (int i = 0; i < 8; ++i) acc[i] = (v8f){};

    const int nChunks = Nn * 8;                        // 16B chunks per 64-wide slab
    for (int ks = 0; ks < Kk; ks += 64) {
        // ---- cooperative stage of BT[:, ks..ks+64) into LDS ----
        for (int c = tid; c < nChunks; c += 256) {
            const int n  = c >> 3;
            const int ko = (c & 7) << 3;               // half offset within slab row
#if HAVE_ASYNC_LDS
            __builtin_amdgcn_global_load_async_to_lds_b128(
                (g1_v4i_p)(BT + (size_t)n * Kk + ks + ko),
                (l3_v4i_p)(ldsB + (size_t)n * 64 + ko), 0, 0);
#else
            *(float4*)(ldsB + (size_t)n * 64 + ko) =
                *(const float4*)(BT + (size_t)n * Kk + ks + ko);
#endif
        }
#if HAVE_ASYNC_LDS
        __builtin_amdgcn_s_wait_asynccnt(0);
#endif
        __syncthreads();

        if (active) {
#pragma unroll
            for (int kh = 0; kh < 2; ++kh) {
                const int k0 = kh * 32;                // local k within slab
                v16h bf[2];
#pragma unroll
                for (int ni = 0; ni < 2; ++ni) {
                    const unsigned int* b0 =
                        (const unsigned int*)(ldsB + (size_t)(tn32 + ni * 16 + l15) * 64)
                        + (k0 >> 1);
                    v8u bu;
#pragma unroll
                    for (int p = 0; p < 4; ++p) { bu[p] = b0[p + hi]; bu[p + 4] = b0[p + 8 + hi]; }
                    bf[ni] = __builtin_bit_cast(v16h, bu);
                }
#pragma unroll
                for (int mi = 0; mi < 4; ++mi) {
                    const unsigned int* a0 = apt[mi] + ((ks + k0) >> 1);
                    __builtin_prefetch(a0 + 32, 0, 1); // global_prefetch next K slab
                    v8u au;
#pragma unroll
                    for (int p = 0; p < 4; ++p) { au[p] = a0[p + hi]; au[p + 4] = a0[p + 8 + hi]; }
                    v16h af = __builtin_bit_cast(v16h, au);
                    acc[mi * 2 + 0] = __builtin_amdgcn_wmma_f32_16x16x32_f16(
                        false, af, false, bf[0], (short)0, acc[mi * 2 + 0], false, false);
                    acc[mi * 2 + 1] = __builtin_amdgcn_wmma_f32_16x16x32_f16(
                        false, af, false, bf[1], (short)0, acc[mi * 2 + 1], false, false);
                }
            }
        }
        __syncthreads();
    }
    if (!active) return;

    // D layout: lane 0-15 -> N=lane, M=r ; lane 16-31 -> N=lane-16, M=r+8
#pragma unroll
    for (int mi = 0; mi < 4; ++mi) {
#pragma unroll
        for (int ni = 0; ni < 2; ++ni) {
            const int nOut  = tn32 + ni * 16 + l15;
            const int mBase = tm64 + mi * 16 + ((lane >> 4) << 3);
            const float bv  = bias[nOut];
            const v8f a = acc[mi * 2 + ni];
#pragma unroll
            for (int r = 0; r < 8; ++r) {
                const int m = mBase + r;
                if (m >= M) continue;
                float v = a[r] + bv;
                if (EPI == 1) v += resid[(size_t)m * Nn + nOut];
                if (EPI == 2) {
                    const float t = 0.7978845608028654f * (v + 0.044715f * v * v * v);
                    Ch[(size_t)m * Nn + nOut] = (_Float16)(0.5f * v * (1.0f + tanhf(t)));
                } else {
                    Cf[(size_t)m * Nn + nOut] = v;
                }
            }
        }
    }
}

// ---------------- FoV mask / stable compaction ----------------
__global__ void mask_build(const int* __restrict__ vp, unsigned char* __restrict__ mask)
{
    int i = blockIdx.x * blockDim.x + threadIdx.x;
    if (i >= NVOX) return;
    int x = vp[3 * i], y = vp[3 * i + 1], z = vp[3 * i + 2];
    if (x >= 0 && x < SX && y >= 0 && y < SY && z >= 0 && z < SZ)
        mask[x * (SY * SZ) + y * SZ + z] = 1;
}

__global__ void mask_count(const unsigned char* __restrict__ mask, int* __restrict__ blkcnt)
{
    __shared__ int cnt;
    if (threadIdx.x == 0) cnt = 0;
    __syncthreads();
    int base = blockIdx.x * 1024 + threadIdx.x * 4, c = 0;
#pragma unroll
    for (int j = 0; j < 4; ++j) c += mask[base + j];
    atomicAdd(&cnt, c);
    __syncthreads();
    if (threadIdx.x == 0) blkcnt[blockIdx.x] = cnt;
}

__global__ void scan_blocks(const int* __restrict__ blkcnt, int* __restrict__ blkoff)
{
    if (threadIdx.x == 0) {
        int s = 0;
        for (int b = 0; b < 256; ++b) { blkoff[b] = s; s += blkcnt[b]; }
        blkoff[256] = s;
    }
}

__global__ void sel_fill(int* __restrict__ sel)
{
    int i = blockIdx.x * blockDim.x + threadIdx.x;
    if (i < KMAX) sel[i] = NVOX;
}

__global__ void sel_write(const unsigned char* __restrict__ mask,
                          const int* __restrict__ blkoff, int* __restrict__ sel)
{
    if (threadIdx.x != 0) return;
    int o = blkoff[blockIdx.x];
    int base = blockIdx.x * 1024;
    for (int j = 0; j < 1024; ++j)
        if (mask[base + j]) { if (o < KMAX) sel[o] = base + j; ++o; }
}

// ---------------- gathers / converts ----------------
__global__ void gather_q(const int* __restrict__ sel,
                         const float* __restrict__ se, const float* __restrict__ sp,
                         const float* __restrict__ rp,
                         float* __restrict__ q, float* __restrict__ qpos,
                         float* __restrict__ ref)
{
    int t = blockIdx.x * blockDim.x + threadIdx.x;
    if (t >= KMAX * EMBED) return;
    int i = t >> 7, c = t & 127;
    int g = sel[i]; if (g > NVOX - 1) g = NVOX - 1;
    q[t]    = se[(size_t)g * EMBED + c];
    qpos[t] = sp[(size_t)g * EMBED + c];
    if (c < 2) ref[i * 2 + c] = rp[(size_t)g * 2 + c];
}

__global__ void feat_to_f16(const float* __restrict__ f0, const float* __restrict__ f1,
                            const float* __restrict__ f2, _Float16* __restrict__ out)
{
    int t = blockIdx.x * blockDim.x + threadIdx.x;
    if (t >= VROWS * EMBED) return;
    int r = t >> 7, c = t & 127;
    float v;
    if (r < HW0)            v = f0[(size_t)c * HW0 + r];
    else if (r < HW0 + HW1) v = f1[(size_t)c * HW1 + (r - HW0)];
    else                    v = f2[(size_t)c * HW2 + (r - HW0 - HW1)];
    out[t] = (_Float16)v;
}

// w f32 [K,N] -> wT f16 [N,K]
__global__ void weight_t_f16(const float* __restrict__ w, _Float16* __restrict__ wt,
                             int Kk, int Nn)
{
    int t = blockIdx.x * blockDim.x + threadIdx.x;
    if (t >= Kk * Nn) return;
    int k = t / Nn, n = t - k * Nn;
    wt[(size_t)n * Kk + k] = (_Float16)w[t];
}

__global__ void add_to_f16(const float* __restrict__ a, const float* __restrict__ b,
                           _Float16* __restrict__ o, int n)
{
    int t = blockIdx.x * blockDim.x + threadIdx.x;
    if (t < n) o[t] = (_Float16)(a[t] + b[t]);
}

__global__ void cvt_f16(const float* __restrict__ a, _Float16* __restrict__ o, int n)
{
    int t = blockIdx.x * blockDim.x + threadIdx.x;
    if (t < n) o[t] = (_Float16)a[t];
}

// ---------------- attention pieces ----------------
__global__ void softmax12(float* __restrict__ aw)
{
    int t = blockIdx.x * blockDim.x + threadIdx.x;   // K*HEADS
    if (t >= KMAX * HEADS) return;
    float* a = aw + (size_t)(t >> 3) * (HEADS * 12) + (t & 7) * 12;
    float mx = a[0];
#pragma unroll
    for (int j = 1; j < 12; ++j) mx = fmaxf(mx, a[j]);
    float e[12], s = 0.f;
#pragma unroll
    for (int j = 0; j < 12; ++j) { e[j] = __expf(a[j] - mx); s += e[j]; }
    float inv = 1.f / s;
#pragma unroll
    for (int j = 0; j < 12; ++j) a[j] = e[j] * inv;
}

__global__ void deform_sample(const float* __restrict__ value,  // [VROWS,128]
                              const float* __restrict__ offb,   // [K,192]
                              const float* __restrict__ awb,    // [K,96] softmaxed
                              const float* __restrict__ ref,    // [K,2]
                              _Float16* __restrict__ outh)      // [K,128] f16
{
    int t = blockIdx.x * blockDim.x + threadIdx.x;
    if (t >= KMAX * EMBED) return;
    const int ch = t & 15, head = (t >> 4) & 7, i = t >> 7;
    const float rx = ref[i * 2], ry = ref[i * 2 + 1];
    const float* off = offb + (size_t)i * 192 + head * (LEVELS * POINTS * 2);
    const float* aw  = awb  + (size_t)i * 96  + head * (LEVELS * POINTS);
    const int FW[3] = {FW0, FW1, FW2};
    const int FHt[3] = {FH0, FH1, FH2};
    float acc = 0.f;
    int start = 0;
#pragma unroll
    for (int l = 0; l < LEVELS; ++l) {
        const int W = FW[l], H = FHt[l];
        const float fWi = 1.f / (float)W, fHi = 1.f / (float)H;
#pragma unroll
        for (int p = 0; p < POINTS; ++p) {
            const float lx = rx + off[(l * POINTS + p) * 2 + 0] * fWi;
            const float ly = ry + off[(l * POINTS + p) * 2 + 1] * fHi;
            const float x = lx * W - 0.5f, y = ly * H - 0.5f;
            const float x0 = floorf(x), y0 = floorf(y);
            const float fx = x - x0, fy = y - y0;
            const int xi = (int)x0, yi = (int)y0;
            float s = 0.f;
#pragma unroll
            for (int dy = 0; dy < 2; ++dy)
#pragma unroll
                for (int dx = 0; dx < 2; ++dx) {
                    const int xx = xi + dx, yy = yi + dy;
                    if (xx >= 0 && xx < W && yy >= 0 && yy < H) {
                        const float w = (dx ? fx : 1.f - fx) * (dy ? fy : 1.f - fy);
                        s += w * value[(size_t)(start + yy * W + xx) * EMBED + head * 16 + ch];
                    }
                }
            acc += aw[l * POINTS + p] * s;
        }
        start += W * H;
    }
    outh[(size_t)i * EMBED + head * 16 + ch] = (_Float16)acc;
}

// wave-per-row LayerNorm over 128 channels (wave32, 4 ch per lane)
__global__ __launch_bounds__(256) void layernorm128(
    const float* __restrict__ x, const float* __restrict__ g,
    const float* __restrict__ b, float* __restrict__ y, int M)
{
    const int lane = threadIdx.x & 31;
    const int row  = (blockIdx.x * blockDim.x + threadIdx.x) >> 5;
    if (row >= M) return;
    const float4 xv = ((const float4*)(x + (size_t)row * EMBED))[lane];
    float s = xv.x + xv.y + xv.z + xv.w;
#pragma unroll
    for (int o = 16; o > 0; o >>= 1) s += __shfl_xor(s, o, 32);
    const float m = s * (1.f / 128.f);
    const float d0 = xv.x - m, d1 = xv.y - m, d2 = xv.z - m, d3 = xv.w - m;
    float vs = d0 * d0 + d1 * d1 + d2 * d2 + d3 * d3;
#pragma unroll
    for (int o = 16; o > 0; o >>= 1) vs += __shfl_xor(vs, o, 32);
    const float inv = rsqrtf(vs * (1.f / 128.f) + 1e-5f);
    const float4 gv = ((const float4*)g)[lane];
    const float4 bv = ((const float4*)b)[lane];
    float4 yv;
    yv.x = d0 * inv * gv.x + bv.x; yv.y = d1 * inv * gv.y + bv.y;
    yv.z = d2 * inv * gv.z + bv.z; yv.w = d3 * inv * gv.w + bv.w;
    ((float4*)(y + (size_t)row * EMBED))[lane] = yv;
}

// ---------------- output assembly ----------------
__global__ void out_transpose(const float* __restrict__ se, float* __restrict__ out)
{
    long long t = (long long)blockIdx.x * blockDim.x + threadIdx.x;
    if (t >= (long long)NVOX * EMBED) return;
    const int n = (int)(t & (NVOX - 1));
    const int c = (int)(t >> 18);
    out[t] = se[(size_t)n * EMBED + c];
}

__global__ void out_scatter(const int* __restrict__ sel, const float* __restrict__ q,
                            float* __restrict__ out)
{
    int t = blockIdx.x * blockDim.x + threadIdx.x;
    if (t >= KMAX * EMBED) return;
    const int i = t >> 7, c = t & 127;
    const int s = sel[i];
    if (s < NVOX) out[(size_t)c * NVOX + s] = q[t];
}

// =====================================================================
extern "C" void kernel_launch(void* const* d_in, const int* in_sizes, int n_in,
                              void* d_out, int out_size, void* d_ws, size_t ws_size,
                              hipStream_t stream)
{
    const float* scene_embed = (const float*)d_in[0];
    const float* feat0       = (const float*)d_in[1];
    const float* feat1       = (const float*)d_in[2];
    const float* feat2       = (const float*)d_in[3];
    const float* scene_pos   = (const float*)d_in[4];
    const int*   vol_pts     = (const int*)d_in[5];
    const float* ref_pix     = (const float*)d_in[6];
    const float* w_v    = (const float*)d_in[7];
    const float* b_v    = (const float*)d_in[8];
    const float* w_off  = (const float*)d_in[9];
    const float* b_off  = (const float*)d_in[10];
    const float* w_attn = (const float*)d_in[11];
    const float* b_attn = (const float*)d_in[12];
    const float* w_out  = (const float*)d_in[13];
    const float* b_out  = (const float*)d_in[14];
    const float* ln1_g  = (const float*)d_in[15];
    const float* ln1_b  = (const float*)d_in[16];
    const float* w_f1   = (const float*)d_in[17];
    const float* b_f1   = (const float*)d_in[18];
    const float* w_f2   = (const float*)d_in[19];
    const float* b_f2   = (const float*)d_in[20];
    const float* ln2_g  = (const float*)d_in[21];
    const float* ln2_b  = (const float*)d_in[22];
    float* out = (float*)d_out;

    // ---- workspace carve ----
    char* ws = (char*)d_ws;
    size_t off = 0;
    auto carve = [&](size_t bytes) -> char* {
        char* p = ws + off;
        off = (off + bytes + 255) & ~(size_t)255;
        return p;
    };
    unsigned char* mask = (unsigned char*)carve(NVOX);
    int* blkcnt = (int*)carve(256 * 4);
    int* blkoff = (int*)carve(257 * 4);
    int* sel    = (int*)carve(KMAX * 4);
    _Float16* featF16 = (_Float16*)carve((size_t)VROWS * EMBED * 2);
    _Float16* wvT   = (_Float16*)carve((size_t)EMBED * EMBED * 2);
    _Float16* woffT = (_Float16*)carve((size_t)192 * EMBED * 2);
    _Float16* wattT = (_Float16*)carve((size_t)96 * EMBED * 2);
    _Float16* woutT = (_Float16*)carve((size_t)EMBED * EMBED * 2);
    _Float16* wf1T  = (_Float16*)carve((size_t)FFN * EMBED * 2);
    _Float16* wf2T  = (_Float16*)carve((size_t)EMBED * FFN * 2);
    float* value = (float*)carve((size_t)VROWS * EMBED * 4);
    float* q     = (float*)carve((size_t)KMAX * EMBED * 4);
    float* qpos  = (float*)carve((size_t)KMAX * EMBED * 4);
    float* ref   = (float*)carve((size_t)KMAX * 2 * 4);
    float* q1    = (float*)carve((size_t)KMAX * EMBED * 4);
    float* offb  = (float*)carve((size_t)KMAX * 192 * 4);   // also reused as [K,128] f32 tmp
    float* awb   = (float*)carve((size_t)KMAX * 96 * 4);
    _Float16* aF16  = (_Float16*)carve((size_t)KMAX * EMBED * 2);  // qp / samp / q1 (f16)
    _Float16* hF16  = (_Float16*)carve((size_t)KMAX * FFN * 2);    // GELU hidden (f16)
    float* tmp = offb;  // [K,128] f32 scratch (fits inside offb)

    const int TB = 256;
    auto blocks = [&](long long n) { return (unsigned)((n + TB - 1) / TB); };
    auto gemmGrid = [&](int M, int N) {
        return (unsigned)((((M + 63) >> 6) * (N >> 5) + 7) / 8);
    };
    auto gemmLds = [&](int N) { return (size_t)N * 64 * sizeof(_Float16); };

    // ---- 1. FoV selection (stable compaction, ascending like jnp.nonzero) ----
    (void)hipMemsetAsync(mask, 0, NVOX, stream);
    mask_build<<<blocks(NVOX), TB, 0, stream>>>(vol_pts, mask);
    mask_count<<<256, 256, 0, stream>>>(mask, blkcnt);
    scan_blocks<<<1, 32, 0, stream>>>(blkcnt, blkoff);
    sel_fill<<<blocks(KMAX), TB, 0, stream>>>(sel);
    sel_write<<<256, 32, 0, stream>>>(mask, blkoff, sel);

    // ---- 2. gathers + f16 staging ----
    gather_q<<<blocks((long long)KMAX * EMBED), TB, 0, stream>>>(
        sel, scene_embed, scene_pos, ref_pix, q, qpos, ref);
    feat_to_f16<<<blocks((long long)VROWS * EMBED), TB, 0, stream>>>(feat0, feat1, feat2, featF16);
    weight_t_f16<<<blocks(EMBED * EMBED), TB, 0, stream>>>(w_v, wvT, EMBED, EMBED);
    weight_t_f16<<<blocks(EMBED * 192), TB, 0, stream>>>(w_off, woffT, EMBED, 192);
    weight_t_f16<<<blocks(EMBED * 96), TB, 0, stream>>>(w_attn, wattT, EMBED, 96);
    weight_t_f16<<<blocks(EMBED * EMBED), TB, 0, stream>>>(w_out, woutT, EMBED, EMBED);
    weight_t_f16<<<blocks(EMBED * FFN), TB, 0, stream>>>(w_f1, wf1T, EMBED, FFN);
    weight_t_f16<<<blocks(FFN * EMBED), TB, 0, stream>>>(w_f2, wf2T, FFN, EMBED);

    // ---- 3. value projection (WMMA) ----
    gemm_wmma_f16<0><<<gemmGrid(VROWS, EMBED), 256, gemmLds(EMBED), stream>>>(
        featF16, wvT, b_v, nullptr, value, nullptr, VROWS, EMBED, EMBED);

    // ---- 4. three deformable layers ----
    for (int layer = 0; layer < LAYERS; ++layer) {
        const int KE = KMAX * EMBED;
        add_to_f16<<<blocks(KE), TB, 0, stream>>>(q, qpos, aF16, KE);          // qp -> f16
        gemm_wmma_f16<0><<<gemmGrid(KMAX, 192), 256, gemmLds(192), stream>>>(
            aF16, woffT, b_off, nullptr, offb, nullptr, KMAX, 192, EMBED);     // offsets
        gemm_wmma_f16<0><<<gemmGrid(KMAX, 96), 256, gemmLds(96), stream>>>(
            aF16, wattT, b_attn, nullptr, awb, nullptr, KMAX, 96, EMBED);      // attn logits
        softmax12<<<blocks(KMAX * HEADS), TB, 0, stream>>>(awb);
        deform_sample<<<blocks(KE), TB, 0, stream>>>(value, offb, awb, ref, aF16);
        gemm_wmma_f16<1><<<gemmGrid(KMAX, EMBED), 256, gemmLds(EMBED), stream>>>(
            aF16, woutT, b_out, q, tmp, nullptr, KMAX, EMBED, EMBED);          // out-proj + resid q
        layernorm128<<<blocks((long long)KMAX * 32), 256, 0, stream>>>(tmp, ln1_g, ln1_b, q1, KMAX);
        cvt_f16<<<blocks(KE), TB, 0, stream>>>(q1, aF16, KE);
        gemm_wmma_f16<2><<<gemmGrid(KMAX, FFN), 256, gemmLds(FFN), stream>>>(
            aF16, wf1T, b_f1, nullptr, nullptr, hF16, KMAX, FFN, EMBED);       // FFN1 + GELU -> f16
        gemm_wmma_f16<1><<<gemmGrid(KMAX, EMBED), 256, gemmLds(EMBED), stream>>>(
            hF16, wf2T, b_f2, q1, tmp, nullptr, KMAX, EMBED, FFN);             // FFN2 + resid q1
        layernorm128<<<blocks((long long)KMAX * 32), 256, 0, stream>>>(tmp, ln2_g, ln2_b, q, KMAX);
    }

    // ---- 5. write voxel grid (transposed) and scatter refined queries ----
    out_transpose<<<blocks((long long)NVOX * EMBED), TB, 0, stream>>>(scene_embed, out);
    out_scatter<<<blocks((long long)KMAX * EMBED), TB, 0, stream>>>(sel, q, out);
    (void)in_sizes; (void)n_in; (void)out_size; (void)ws_size;
}
